// StackedContextViT_MLA_50251117363438
// MI455X (gfx1250) — matrix operand order repaired and checked by reading, not compiled, not run on hardware
//
#include <hip/hip_runtime.h>

typedef __attribute__((ext_vector_type(16))) __bf16 v16bf;
typedef __attribute__((ext_vector_type(8)))  float  v8f;

#define WMMA_BF16(a, b, c) \
    __builtin_amdgcn_wmma_f32_16x16x32_bf16(false, (a), false, (b), (short)0, (c), false, false)

struct B16x8 { unsigned int w[4]; };   // 8 bf16 = 16 bytes

// ---------------------------------------------------------------------------
// Generic GEMM: C[M,N] = A[M,K] @ B[K,N] + bias[N]
//   HASROW : += addRow[(gm & 1023) * N + gn]   (pos-embedding fuse)
//   GELU   : exact gelu epilogue
//   OUTBF16: store bf16 (for Q/K/V feeding the attention kernel)
// Dims: M%128==0, N%128==0, K%64==0 (all true for this model).
// Tile 128x128x64, 256 threads (8 waves), wave = 32x64 (2x4 WMMA tiles),
// 16 v_wmma per LDS stage, global_prefetch of the next K stage.
// ---------------------------------------------------------------------------
template <bool HASROW, bool GELU, bool OUTBF16>
__global__ __launch_bounds__(256) void gemm_bf16_wmma(
    const float* __restrict__ A, const float* __restrict__ Bm,
    const float* __restrict__ bias, void* __restrict__ Cout,
    int M, int N, int K,
    const float* __restrict__ addRow)
{
    __shared__ __bf16 As[128][64];   // [m][k]  16KB
    __shared__ __bf16 Bs[128][64];   // [n][k]  16KB (transposed on load)

    const int tid  = threadIdx.x;
    const int lane = tid & 31;
    const int wid  = tid >> 5;
    const int wm   = (wid & 3) * 32;
    const int wn   = (wid >> 2) * 64;
    const int bm   = blockIdx.y * 128;
    const int bn   = blockIdx.x * 128;
    const int hrow = lane & 15;
    const int lg   = lane >> 4;

    v8f acc[2][4];
#pragma unroll
    for (int mi = 0; mi < 2; ++mi)
#pragma unroll
        for (int ni = 0; ni < 4; ++ni)
#pragma unroll
            for (int e = 0; e < 8; ++e) acc[mi][ni][e] = 0.f;

    for (int k0 = 0; k0 < K; k0 += 64) {
        __syncthreads();
#pragma unroll
        for (int i = 0; i < 8; ++i) {
            const int u = tid + i * 256;            // 2048 float4 units each
            // A tile: 128 rows x 64 cols
            const int m  = u >> 4;
            const int kq = (u & 15) << 2;
            const float* ap = A + (size_t)(bm + m) * K + k0 + kq;
            const float4 fa = *(const float4*)ap;
            As[m][kq + 0] = (__bf16)fa.x; As[m][kq + 1] = (__bf16)fa.y;
            As[m][kq + 2] = (__bf16)fa.z; As[m][kq + 3] = (__bf16)fa.w;
            // B tile: 64 rows(k) x 128 cols(n), store transposed
            const int kr = u >> 5;
            const int nq = (u & 31) << 2;
            const float* bp = Bm + (size_t)(k0 + kr) * N + bn + nq;
            const float4 fb = *(const float4*)bp;
            Bs[nq + 0][kr] = (__bf16)fb.x; Bs[nq + 1][kr] = (__bf16)fb.y;
            Bs[nq + 2][kr] = (__bf16)fb.z; Bs[nq + 3][kr] = (__bf16)fb.w;
            // Prefetch next K stage into cache while this stage computes
            if (k0 + 64 < K) {
                __builtin_prefetch(ap + 64, 0, 1);
                __builtin_prefetch(bp + (size_t)64 * N, 0, 1);
            }
        }
        __syncthreads();

#pragma unroll
        for (int ks = 0; ks < 2; ++ks) {
            const int ko = ks * 32;
            v16bf af[2], bf[4];
#pragma unroll
            for (int mi = 0; mi < 2; ++mi) {
                const __bf16* p = &As[wm + mi * 16 + hrow][ko + lg * 8];
#pragma unroll
                for (int j = 0; j < 8; ++j) { af[mi][j] = p[j]; af[mi][j + 8] = p[j + 16]; }
            }
#pragma unroll
            for (int ni = 0; ni < 4; ++ni) {
                const __bf16* p = &Bs[wn + ni * 16 + hrow][ko + lg * 8];
#pragma unroll
                for (int j = 0; j < 8; ++j) { bf[ni][j] = p[j]; bf[ni][j + 8] = p[j + 16]; }
            }
#pragma unroll
            for (int mi = 0; mi < 2; ++mi)
#pragma unroll
                for (int ni = 0; ni < 4; ++ni)
                    acc[mi][ni] = WMMA_BF16(af[mi], bf[ni], acc[mi][ni]);
        }
    }

#pragma unroll
    for (int mi = 0; mi < 2; ++mi)
#pragma unroll
        for (int ni = 0; ni < 4; ++ni) {
            const int gn = bn + wn + ni * 16 + hrow;
            const float bv = bias[gn];
#pragma unroll
            for (int e = 0; e < 8; ++e) {
                const int gm = bm + wm + mi * 16 + e + lg * 8;
                float v = acc[mi][ni][e] + bv;
                if (HASROW) v += addRow[(size_t)(gm & 1023) * N + gn];
                if (GELU)   v = 0.5f * v * (1.f + erff(v * 0.70710678118f));
                if (OUTBF16) ((__bf16*)Cout)[(size_t)gm * N + gn] = (__bf16)v;
                else         ((float*)Cout)[(size_t)gm * N + gn] = v;
            }
        }
}

// ---------------------------------------------------------------------------
// Fused flash attention over bf16 Q/K/V: grid = B*H*(N/128), 256 thr (8 waves).
// Each wave: 16 q rows; workgroup stages 32-key K/V chunks in LDS.
// hd = 64, scale = 1/8, rel_bias[N,N] added to scores. Output f32.
// ---------------------------------------------------------------------------
__global__ __launch_bounds__(256) void attn_flash_wmma(
    const __bf16* __restrict__ q, const __bf16* __restrict__ k,
    const __bf16* __restrict__ v, const float* __restrict__ rel,
    float* __restrict__ o)
{
    __shared__ __bf16 Kt[32][64];      // [key][hd]
    __shared__ __bf16 Vt[64][32];      // [hd][key]
    __shared__ __bf16 Pl[8][16][32];   // per-wave P relayout scratch

    const int tid  = threadIdx.x;
    const int lane = tid & 31;
    const int wid  = tid >> 5;
    const int qblk = blockIdx.x & 7;
    const int bh   = blockIdx.x >> 3;
    const int b    = bh / 12;
    const int h    = bh % 12;
    const int hrow = lane & 15;
    const int lg   = lane >> 4;
    const int qtokw = qblk * 128 + wid * 16;     // wave's first q token
    const float scale = 0.125f;                  // 1/sqrt(64)

    // Load Q fragments (A layout, 16x64 split into two 16x32 frags), bf16 direct
    v16bf qf[2];
    {
        const __bf16* qp = q + (size_t)(b * 1024 + qtokw + hrow) * 768 + h * 64;
#pragma unroll
        for (int f = 0; f < 2; ++f) {
            const int base = f * 32 + lg * 8;
#pragma unroll
            for (int j = 0; j < 8; ++j) {
                qf[f][j]     = qp[base + j];
                qf[f][j + 8] = qp[base + 16 + j];
            }
        }
    }

    v8f oacc[4];
#pragma unroll
    for (int ni = 0; ni < 4; ++ni)
#pragma unroll
        for (int e = 0; e < 8; ++e) oacc[ni][e] = 0.f;
    float mstat[8], lstat[8];
#pragma unroll
    for (int e = 0; e < 8; ++e) { mstat[e] = -3.0e38f; lstat[e] = 0.f; }

    // Per-thread staging coordinates: one 16B chunk per matrix per chunk-stage
    const int skey = tid >> 3;             // 0..31
    const int soff = (tid & 7) << 3;       // 0..56, bf16 elems

    for (int kb = 0; kb < 1024; kb += 32) {
        __syncthreads();
        {
            const size_t row = (size_t)(b * 1024 + kb + skey) * 768 + h * 64 + soff;
            // K chunk: straight 16B copy, [key][hd]
            *(B16x8*)&Kt[skey][soff] = *(const B16x8*)(k + row);
            // V chunk: 16B load, transpose-store to [hd][key]
            const __bf16* vp = v + row;
#pragma unroll
            for (int c = 0; c < 8; ++c) Vt[soff + c][skey] = vp[c];
            if (kb + 32 < 1024) {
                __builtin_prefetch(k + row + 32 * 768, 0, 1);
                __builtin_prefetch(v + row + 32 * 768, 0, 1);
            }
        }
        __syncthreads();

        // Scores: two 16x16 tiles over this 32-key chunk, contraction hd=64
        v8f s[2];
#pragma unroll
        for (int nt = 0; nt < 2; ++nt) {
#pragma unroll
            for (int e = 0; e < 8; ++e) s[nt][e] = 0.f;
#pragma unroll
            for (int f = 0; f < 2; ++f) {
                v16bf bfr;
                const __bf16* kp = &Kt[nt * 16 + hrow][f * 32 + lg * 8];
#pragma unroll
                for (int j = 0; j < 8; ++j) { bfr[j] = kp[j]; bfr[j + 8] = kp[j + 16]; }
                s[nt] = WMMA_BF16(qf[f], bfr, s[nt]);
            }
        }

        // Online softmax per row (row = vg + lg*8 of this wave's 16 q rows)
        float p0[8], p1[8], alpha[8];
#pragma unroll
        for (int vg = 0; vg < 8; ++vg) {
            const int qtok = qtokw + vg + lg * 8;
            const float* rp = rel + (size_t)qtok * 1024 + kb;
            float s0 = s[0][vg] * scale + rp[hrow];
            float s1 = s[1][vg] * scale + rp[16 + hrow];
            float mx = fmaxf(s0, s1);
            for (int off = 1; off < 16; off <<= 1)
                mx = fmaxf(mx, __shfl_xor(mx, off, 32));
            const float mnew = fmaxf(mstat[vg], mx);
            const float a = __expf(mstat[vg] - mnew);
            const float e0 = __expf(s0 - mnew);
            const float e1 = __expf(s1 - mnew);
            float rs = e0 + e1;
            for (int off = 1; off < 16; off <<= 1)
                rs += __shfl_xor(rs, off, 32);
            lstat[vg] = lstat[vg] * a + rs;
            mstat[vg] = mnew;
            alpha[vg] = a;
            p0[vg] = e0; p1[vg] = e1;
        }
#pragma unroll
        for (int ni = 0; ni < 4; ++ni)
#pragma unroll
            for (int vg = 0; vg < 8; ++vg) oacc[ni][vg] *= alpha[vg];

        // Re-layout P (C-frag -> A-frag) through wave-private LDS
#pragma unroll
        for (int vg = 0; vg < 8; ++vg) {
            const int r = vg + lg * 8;
            Pl[wid][r][hrow]      = (__bf16)p0[vg];
            Pl[wid][r][16 + hrow] = (__bf16)p1[vg];
        }
        v16bf pf;
        {
            const __bf16* pp = &Pl[wid][hrow][lg * 8];
#pragma unroll
            for (int j = 0; j < 8; ++j) { pf[j] = pp[j]; pf[j + 8] = pp[j + 16]; }
        }
        // O += P @ V  (contraction over 32 keys, 4 hd tiles)
#pragma unroll
        for (int ni = 0; ni < 4; ++ni) {
            v16bf bfr;
            const __bf16* vp2 = &Vt[ni * 16 + hrow][lg * 8];
#pragma unroll
            for (int j = 0; j < 8; ++j) { bfr[j] = vp2[j]; bfr[j + 8] = vp2[j + 16]; }
            oacc[ni] = WMMA_BF16(pf, bfr, oacc[ni]);
        }
    }

    float rinv[8];
#pragma unroll
    for (int vg = 0; vg < 8; ++vg) rinv[vg] = 1.f / lstat[vg];
#pragma unroll
    for (int ni = 0; ni < 4; ++ni)
#pragma unroll
        for (int vg = 0; vg < 8; ++vg) {
            const int qtok = qtokw + vg + lg * 8;
            o[(size_t)(b * 1024 + qtok) * 768 + h * 64 + ni * 16 + hrow] =
                oacc[ni][vg] * rinv[vg];
        }
}

// ---------------------------------------------------------------------------
// Fused residual + LayerNorm: out = LN(xin + dlt) * g + beta. One block/row.
// ---------------------------------------------------------------------------
__global__ __launch_bounds__(256) void ln_residual(
    const float* __restrict__ xin, const float* __restrict__ dlt,
    const float* __restrict__ g, const float* __restrict__ beta,
    float* __restrict__ out)
{
    __shared__ float sb[256];
    const int row = blockIdx.x;
    const int tid = threadIdx.x;
    float r[3];
    float s = 0.f;
#pragma unroll
    for (int j = 0; j < 3; ++j) {
        const int c = tid + j * 256;
        r[j] = xin[(size_t)row * 768 + c] + dlt[(size_t)row * 768 + c];
        s += r[j];
    }
    sb[tid] = s; __syncthreads();
    for (int st = 128; st > 0; st >>= 1) { if (tid < st) sb[tid] += sb[tid + st]; __syncthreads(); }
    const float mean = sb[0] * (1.f / 768.f);
    __syncthreads();
    float ss = 0.f;
#pragma unroll
    for (int j = 0; j < 3; ++j) { const float t = r[j] - mean; ss += t * t; }
    sb[tid] = ss; __syncthreads();
    for (int st = 128; st > 0; st >>= 1) { if (tid < st) sb[tid] += sb[tid + st]; __syncthreads(); }
    const float rstd = rsqrtf(sb[0] * (1.f / 768.f) + 1e-5f);
#pragma unroll
    for (int j = 0; j < 3; ++j) {
        const int c = tid + j * 256;
        out[(size_t)row * 768 + c] = (r[j] - mean) * rstd * g[c] + beta[c];
    }
}

__global__ void mask_embed(const float* __restrict__ patches,
                           const unsigned char* __restrict__ mask,
                           const float* __restrict__ token,
                           float* __restrict__ out, int total)
{
    const int i = blockIdx.x * 256 + threadIdx.x;
    if (i >= total) return;
    const int row = i >> 8, col = i & 255;
    out[i] = mask[row] ? token[col] : patches[i];
}

__global__ void mask_passthrough(const unsigned char* __restrict__ mask,
                                 float* __restrict__ out, int n)
{
    const int i = blockIdx.x * 256 + threadIdx.x;
    if (i < n) out[i] = mask[i] ? 1.f : 0.f;
}

// ---------------------------------------------------------------------------
extern "C" void kernel_launch(void* const* d_in, const int* in_sizes, int n_in,
                              void* d_out, int out_size, void* d_ws, size_t ws_size,
                              hipStream_t stream)
{
    (void)in_sizes; (void)n_in; (void)ws_size;
    const float* patches   = (const float*)d_in[0];
    const unsigned char* mask = (const unsigned char*)d_in[1];
    const float* mask_tok  = (const float*)d_in[2];
    const float* Wp        = (const float*)d_in[3];
    const float* bp        = (const float*)d_in[4];
    const float* pos_emb   = (const float*)d_in[5];
    const float* rel_bias  = (const float*)d_in[6];
    const float* Wq        = (const float*)d_in[7];
    const float* bq        = (const float*)d_in[8];
    const float* WaK       = (const float*)d_in[9];
    const float* baK       = (const float*)d_in[10];
    const float* WbK       = (const float*)d_in[11];
    const float* bbK       = (const float*)d_in[12];
    const float* WaV       = (const float*)d_in[13];
    const float* baV       = (const float*)d_in[14];
    const float* WbV       = (const float*)d_in[15];
    const float* bbV       = (const float*)d_in[16];
    const float* Wo        = (const float*)d_in[17];
    const float* bo        = (const float*)d_in[18];
    const float* W1        = (const float*)d_in[19];
    const float* b1        = (const float*)d_in[20];
    const float* W2        = (const float*)d_in[21];
    const float* b2        = (const float*)d_in[22];
    const float* g1        = (const float*)d_in[23];
    const float* beta1     = (const float*)d_in[24];
    const float* g2        = (const float*)d_in[25];
    const float* beta2     = (const float*)d_in[26];
    const float* Wout      = (const float*)d_in[27];
    const float* bout      = (const float*)d_in[28];

    const int M = 4096, E = 768, Lt = 256, F = 3072, Vv = 512;
    const size_t SX = (size_t)M * E;
    float* ws = (float*)d_ws;
    float*  x  = ws;
    __bf16* qb = (__bf16*)(ws + 1 * SX);     // bf16 Q (uses half the slot)
    __bf16* kb = (__bf16*)(ws + 2 * SX);     // bf16 K
    __bf16* vb = (__bf16*)(ws + 3 * SX);     // bf16 V
    float*  ob = ws + 4 * SX;
    float*  tb = ws + 5 * SX;                // up to 4096*3072 floats

    const dim3 blk(256);

    // Embedding: masked patches -> tb, then tb @ Wp + bp + pos_emb -> x
    mask_embed<<<(M * 256) / 256, blk, 0, stream>>>(patches, mask, mask_tok, tb, M * 256);
    gemm_bf16_wmma<true, false, false><<<dim3(E / 128, M / 128), blk, 0, stream>>>(
        tb, Wp, bp, x, M, E, 256, pos_emb);

    for (int i = 0; i < 4; ++i) {
        // Q (bf16 out)
        gemm_bf16_wmma<false, false, true><<<dim3(E / 128, M / 128), blk, 0, stream>>>(
            x, Wq + (size_t)i * E * E, bq + (size_t)i * E, qb, M, E, E, nullptr);
        // K = (x @ WaK + baK) @ WbK + bbK  (second stage bf16 out)
        gemm_bf16_wmma<false, false, false><<<dim3(Lt / 128, M / 128), blk, 0, stream>>>(
            x, WaK + (size_t)i * E * Lt, baK + (size_t)i * Lt, tb, M, Lt, E, nullptr);
        gemm_bf16_wmma<false, false, true><<<dim3(E / 128, M / 128), blk, 0, stream>>>(
            tb, WbK + (size_t)i * Lt * E, bbK + (size_t)i * E, kb, M, E, Lt, nullptr);
        // V = (x @ WaV + baV) @ WbV + bbV  (second stage bf16 out)
        gemm_bf16_wmma<false, false, false><<<dim3(Lt / 128, M / 128), blk, 0, stream>>>(
            x, WaV + (size_t)i * E * Lt, baV + (size_t)i * Lt, tb, M, Lt, E, nullptr);
        gemm_bf16_wmma<false, false, true><<<dim3(E / 128, M / 128), blk, 0, stream>>>(
            tb, WbV + (size_t)i * Lt * E, bbV + (size_t)i * E, vb, M, E, Lt, nullptr);
        // Attention (fused flash, scale + rel_bias inside), f32 out
        attn_flash_wmma<<<4 * 12 * 8, blk, 0, stream>>>(qb, kb, vb, rel_bias, ob);
        // O projection
        gemm_bf16_wmma<false, false, false><<<dim3(E / 128, M / 128), blk, 0, stream>>>(
            ob, Wo + (size_t)i * E * E, bo + (size_t)i * E, tb, M, E, E, nullptr);
        // x = LN(x + O)
        ln_residual<<<M, blk, 0, stream>>>(x, tb, g1 + (size_t)i * E, beta1 + (size_t)i * E, x);
        // FFN
        gemm_bf16_wmma<false, true, false><<<dim3(F / 128, M / 128), blk, 0, stream>>>(
            x, W1 + (size_t)i * E * F, b1 + (size_t)i * F, tb, M, F, E, nullptr);
        gemm_bf16_wmma<false, false, false><<<dim3(E / 128, M / 128), blk, 0, stream>>>(
            tb, W2 + (size_t)i * F * E, b2 + (size_t)i * E, ob, M, E, F, nullptr);
        ln_residual<<<M, blk, 0, stream>>>(x, ob, g2 + (size_t)i * E, beta2 + (size_t)i * E, x);
    }

    // Output head -> d_out (logits), plus mask passthrough tail if present
    gemm_bf16_wmma<false, false, false><<<dim3(Vv / 128, M / 128), blk, 0, stream>>>(
        x, Wout, bout, (float*)d_out, M, Vv, E, nullptr);
    const int logits = M * Vv;
    if (out_size >= logits + M)
        mask_passthrough<<<(M + 255) / 256, blk, 0, stream>>>(
            mask, (float*)d_out + logits, M);
}
